// RNNs_81011673137155
// MI455X (gfx1250) — matrix operand order repaired and checked by reading, not compiled, not run on hardware
//
#include <hip/hip_runtime.h>

typedef _Float16 v16h __attribute__((ext_vector_type(16)));
typedef _Float16 v8h  __attribute__((ext_vector_type(8)));
typedef float    v8f  __attribute__((ext_vector_type(8)));

#define WAVES   8
#define THREADS 256
#define TSTEPS  12
#define BATCH   64
#define NNODES  2048
#define NSEQ    (BATCH * NNODES)   // 131072 sequences

// ---- fast transcendental helpers (prefer native v_tanh_f32 on gfx1250) ----
__device__ __forceinline__ float fexp2(float x) {
#if __has_builtin(__builtin_amdgcn_exp2f)
    return __builtin_amdgcn_exp2f(x);       // raw v_exp_f32
#else
    return exp2f(x);
#endif
}
__device__ __forceinline__ float frcp(float x) {
#if __has_builtin(__builtin_amdgcn_rcpf)
    return __builtin_amdgcn_rcpf(x);        // raw v_rcp_f32 (no div-scale fixup)
#else
    return 1.0f / x;
#endif
}
__device__ __forceinline__ float ftanh(float x) {
#if __has_builtin(__builtin_amdgcn_tanhf)
    return __builtin_amdgcn_tanhf(x);       // native v_tanh_f32
#else
    return fmaf(-2.0f, frcp(1.0f + fexp2(2.885390081777927f * x)), 1.0f);
#endif
}
__device__ __forceinline__ float fsig(float x) {
#if __has_builtin(__builtin_amdgcn_tanhf)
    return fmaf(0.5f, __builtin_amdgcn_tanhf(0.5f * x), 0.5f);
#else
    return frcp(1.0f + fexp2(-1.442695040888963f * x));
#endif
}

__device__ __forceinline__ v8f wmma_f16(v16h a, v16h b, v8f c) {
    // v_wmma_f32_16x16x32_f16: D = A(16x32,f16) * B(32x16,f16) + C(16x16,f32)
    return __builtin_amdgcn_wmma_f32_16x16x32_f16(false, a, false, b, (short)0, c, false, false);
}

// B operand: 16 contiguous halves per lane (32B, aligned) -> 2x ds_load_b128
__device__ __forceinline__ v16h ldsB(const _Float16* p) {
    return *(const v16h*)p;
}

// A operand gather from row-major [16][32] f16 staging.
// A layout: M = lane%16; halves j=0..15 -> K = 16*hi + j
__device__ __forceinline__ v16h loadA(const _Float16* row, int hi) {
    v8h l8 = *(const v8h*)(row + 16 * hi);
    v8h h8 = *(const v8h*)(row + 16 * hi + 8);
    return __builtin_shufflevector(l8, h8, 0,1,2,3,4,5,6,7,8,9,10,11,12,13,14,15);
}

__global__ __launch_bounds__(THREADS) void gru_wmma_kernel(
    const float* __restrict__ in,      // [B,1,N,T]  flat: seq*12 + t
    const float* __restrict__ start_w, // [32,1]
    const float* __restrict__ start_b, // [32]
    const float* __restrict__ w1x,     // [32,96]
    const float* __restrict__ w1h,     // [32,96]
    const float* __restrict__ b1x,     // [96]
    const float* __restrict__ b1h,     // [96]
    const float* __restrict__ w2x,     // [32,96]
    const float* __restrict__ w2h,     // [32,96]
    const float* __restrict__ b2x,     // [96]
    const float* __restrict__ b2h,     // [96]
    const float* __restrict__ ew,      // [12,32,12] (o,c,t)
    const float* __restrict__ eb,      // [12]
    float* __restrict__ out)           // [B,1,N,12] flat: seq*12 + o
{
    // ---- LDS ----
    __shared__ __attribute__((aligned(32))) _Float16 wB[3 * 6 * 32 * 16];   // wh1, wx2, wh2 in B-layout
    __shared__ __attribute__((aligned(32))) _Float16 ewB[TSTEPS * 32 * 16]; // end_w B-tiles per t
    __shared__ float Avec[96];    // start_w^T . wx1
    __shared__ float Bcon[96];    // start_b^T . wx1 + bx1
    __shared__ __attribute__((aligned(16))) float u_t[WAVES][TSTEPS][16];   // transposed: [t][m]
    __shared__ __attribute__((aligned(32))) _Float16 hstage[WAVES][2][16 * 32];

    const int tid  = threadIdx.x;
    const int wave = tid >> 5;
    const int lane = tid & 31;
    const int lo   = lane & 15;
    const int hi   = lane >> 4;

    // ---- pack gate weights into WMMA B-layout: lane n = lo, halves j -> K = j + 16*hi ----
    {
        const float* mats[3] = { w1h, w2x, w2h };
        for (int e = tid; e < 3 * 6 * 32 * 16; e += THREADS) {
            int j  = e & 15;
            int ln = (e >> 4) & 31;
            int g  = e >> 9;          // 0..17
            int tl = g % 6;
            int mt = g / 6;
            int k  = j + 16 * (ln >> 4);
            int n  = tl * 16 + (ln & 15);
            wB[e] = (_Float16)mats[mt][k * 96 + n];
        }
        for (int e = tid; e < TSTEPS * 32 * 16; e += THREADS) {
            int j  = e & 15;
            int ln = (e >> 4) & 31;
            int t  = e >> 9;
            int k  = j + 16 * (ln >> 4);
            int o  = ln & 15;
            ewB[e] = (o < 12) ? (_Float16)ew[(o * 32 + k) * 12 + t] : (_Float16)0.0f;
        }
        if (tid < 96) {
            float a = 0.0f, bc = 0.0f;
            for (int c = 0; c < 32; ++c) {
                float w = w1x[c * 96 + tid];
                a  += start_w[c] * w;
                bc += start_b[c] * w;
            }
            Avec[tid] = a;
            Bcon[tid] = bc + b1x[tid];
        }
    }

    // ---- stage per-wave inputs, transposed to [t][m] so per-t reads are contiguous ----
    const int tileBase = (blockIdx.x * WAVES + wave) * 16;
    if (lane < 16) {
        const float4* src = (const float4*)(in + (tileBase + lane) * TSTEPS);
        float4 a = src[0], b = src[1], c = src[2];
        float v[TSTEPS] = { a.x, a.y, a.z, a.w, b.x, b.y, b.z, b.w, c.x, c.y, c.z, c.w };
#pragma unroll
        for (int t = 0; t < TSTEPS; ++t) u_t[wave][t][lane] = v[t];
    }
    __syncthreads();

    // ---- register-cached per-lane constants (column = function of lo only) ----
    float Az[2], Ar[2], An[2], Bz[2], Br[2], Bn[2];
    float b1z[2], b1r[2], b1n[2];
    float bzr[4], bxn[2], bhn[2];
#pragma unroll
    for (int q = 0; q < 2; ++q) {
        int cz = 16 * q + lo, cr = cz + 32, cn = cz + 64;
        Az[q] = Avec[cz]; Ar[q] = Avec[cr]; An[q] = Avec[cn];
        Bz[q] = Bcon[cz]; Br[q] = Bcon[cr]; Bn[q] = Bcon[cn];
        b1z[q] = b1h[cz]; b1r[q] = b1h[cr]; b1n[q] = b1h[cn];
        bxn[q] = b2x[cn]; bhn[q] = b2h[cn];
    }
#pragma unroll
    for (int tl = 0; tl < 4; ++tl) {
        int c = 16 * tl + lo;
        bzr[tl] = b2x[c] + b2h[c];
    }
    const float ebv = (lo < 12) ? eb[lo] : 0.0f;

    // ---- state ----
    v8f h1[2], h2[2], yacc;
    v16h a1, a2;
#pragma unroll
    for (int i = 0; i < 8; ++i) { h1[0][i] = 0.f; h1[1][i] = 0.f; h2[0][i] = 0.f; h2[1][i] = 0.f; yacc[i] = 0.f; }
#pragma unroll
    for (int i = 0; i < 16; ++i) { a1[i] = (_Float16)0.f; a2[i] = (_Float16)0.f; }

    _Float16* st1 = &hstage[wave][0][0];
    _Float16* st2 = &hstage[wave][1][0];

    for (int t = 0; t < TSTEPS; ++t) {
        // u values for this wave's rows m = 8*hi .. 8*hi+7: one 32B contiguous chunk
        const float4* up = (const float4*)&u_t[wave][t][8 * hi];
        float4 ua = up[0], ub = up[1];
        float uv[8] = { ua.x, ua.y, ua.z, ua.w, ub.x, ub.y, ub.z, ub.w };

        // ===== layer 1: gh1 = a1 @ wh1 + bh1  (a1 == h1[t-1] in f16 A-layout) =====
        v8f g1[6];
#pragma unroll
        for (int tl = 0; tl < 6; ++tl) {
            float bias = (tl < 2) ? b1z[tl] : ((tl < 4) ? b1r[tl - 2] : b1n[tl - 4]);
            v8f c;
#pragma unroll
            for (int i = 0; i < 8; ++i) c[i] = bias;
            g1[tl] = wmma_f16(a1, ldsB(&wB[(0 * 6 + tl) * 512]), c);
        }
#pragma unroll
        for (int q = 0; q < 2; ++q)
#pragma unroll
            for (int r = 0; r < 8; ++r) {
                float u  = uv[r];
                float z  = fsig(fmaf(u, Az[q], Bz[q]) + g1[q][r]);
                float rg = fsig(fmaf(u, Ar[q], Br[q]) + g1[2 + q][r]);
                float ng = ftanh(fmaf(u, An[q], Bn[q]) + rg * g1[4 + q][r]);
                h1[q][r] = fmaf(z, h1[q][r] - ng, ng);
            }
        // D-layout f32 -> row-major f16 staging -> A-layout
#pragma unroll
        for (int q = 0; q < 2; ++q)
#pragma unroll
            for (int r = 0; r < 8; ++r)
                st1[(r + 8 * hi) * 32 + 16 * q + lo] = (_Float16)h1[q][r];
        asm volatile("s_wait_dscnt 0" ::: "memory");
        a1 = loadA(st1 + lo * 32, hi);

        // ===== layer 2: z/r gates fuse x@wx2 + h@wh2 into one accumulator =====
        v8f gzr[4], gxn[2], ghn[2];
#pragma unroll
        for (int tl = 0; tl < 4; ++tl) {
            v8f c;
#pragma unroll
            for (int i = 0; i < 8; ++i) c[i] = bzr[tl];
            c       = wmma_f16(a1, ldsB(&wB[(1 * 6 + tl) * 512]), c);
            gzr[tl] = wmma_f16(a2, ldsB(&wB[(2 * 6 + tl) * 512]), c);
        }
#pragma unroll
        for (int q = 0; q < 2; ++q) {
            v8f c, d;
#pragma unroll
            for (int i = 0; i < 8; ++i) { c[i] = bxn[q]; d[i] = bhn[q]; }
            gxn[q] = wmma_f16(a1, ldsB(&wB[(1 * 6 + 4 + q) * 512]), c);
            ghn[q] = wmma_f16(a2, ldsB(&wB[(2 * 6 + 4 + q) * 512]), d);
        }
#pragma unroll
        for (int q = 0; q < 2; ++q)
#pragma unroll
            for (int r = 0; r < 8; ++r) {
                float z  = fsig(gzr[q][r]);
                float rg = fsig(gzr[2 + q][r]);
                float ng = ftanh(fmaf(rg, ghn[q][r], gxn[q][r]));
                h2[q][r] = fmaf(z, h2[q][r] - ng, ng);
            }
#pragma unroll
        for (int q = 0; q < 2; ++q)
#pragma unroll
            for (int r = 0; r < 8; ++r)
                st2[(r + 8 * hi) * 32 + 16 * q + lo] = (_Float16)h2[q][r];
        asm volatile("s_wait_dscnt 0" ::: "memory");
        a2 = loadA(st2 + lo * 32, hi);

        // ===== end-conv accumulate: yacc += h2[t] @ end_w[:,:,t] =====
        yacc = wmma_f16(a2, ldsB(&ewB[t * 512]), yacc);
    }

    // ---- store: D-layout (m = r + 8*hi, o = lo), only o < 12 valid ----
#pragma unroll
    for (int r = 0; r < 8; ++r) {
        if (lo < 12) {
            int m = tileBase + r + 8 * hi;
            out[m * 12 + lo] = yacc[r] + ebv;
        }
    }
}

extern "C" void kernel_launch(void* const* d_in, const int* in_sizes, int n_in,
                              void* d_out, int out_size, void* d_ws, size_t ws_size,
                              hipStream_t stream) {
    const float* in      = (const float*)d_in[0];
    const float* start_w = (const float*)d_in[1];
    const float* start_b = (const float*)d_in[2];
    const float* w1x     = (const float*)d_in[3];
    const float* w1h     = (const float*)d_in[4];
    const float* b1x     = (const float*)d_in[5];
    const float* b1h     = (const float*)d_in[6];
    const float* w2x     = (const float*)d_in[7];
    const float* w2h     = (const float*)d_in[8];
    const float* b2x     = (const float*)d_in[9];
    const float* b2h     = (const float*)d_in[10];
    const float* ew      = (const float*)d_in[11];
    const float* eb      = (const float*)d_in[12];
    float* out           = (float*)d_out;

    // 131072 sequences / (16 per wave * 8 waves per block) = 1024 blocks
    dim3 grid(NSEQ / (16 * WAVES));
    dim3 block(THREADS);
    gru_wmma_kernel<<<grid, block, 0, stream>>>(
        in, start_w, start_b, w1x, w1h, b1x, b1h,
        w2x, w2h, b2x, b2h, ew, eb, out);
}
